// SimpleSemanticModel_78400333021648
// MI455X (gfx1250) — compile-verified, matrix operand order checked by reading, not run on hardware
//
#include <hip/hip_runtime.h>

#define VOCAB 50257
#define NPAD  50304   /* 393 * 128 */
#define KDIM  1024
#define TM    128
#define TN    128
#define TK    32
#define LDSP  40      /* padded LDS row stride in bf16 elems (80B) */

typedef unsigned short ushort_t;
typedef __attribute__((ext_vector_type(16))) __bf16 v16bf;
typedef __attribute__((ext_vector_type(8)))  float  v8f;

union FragBF { v16bf v; uint4 q[2]; };

#if defined(__gfx1250__) && __has_builtin(__builtin_amdgcn_global_load_async_to_lds_b128)
#define ASYNC_COPY 1
#else
#define ASYNC_COPY 0
#endif

__device__ __forceinline__ ushort_t f2bf(float f) {
    unsigned int u = __float_as_uint(f);
    u += 0x7FFFu + ((u >> 16) & 1u);          // round-to-nearest-even
    return (ushort_t)(u >> 16);
}

#if ASYNC_COPY
// Exact param types per clang diagnostic: AS(1)/AS(3) pointers to 'int __vector(4)'
typedef int vsi4 __attribute__((__vector_size__(4 * sizeof(int))));
typedef __attribute__((address_space(1))) vsi4 as1_vsi4;
typedef __attribute__((address_space(3))) vsi4 as3_vsi4;

// 16B per-lane async copy: global -> LDS, bypassing VGPRs (tracked by ASYNCcnt).
// Low 32 bits of a generic LDS pointer are the LDS byte offset (flat aperture rule).
__device__ __forceinline__ void async_copy16(const ushort_t* g, ushort_t* l) {
    __builtin_amdgcn_global_load_async_to_lds_b128(
        (as1_vsi4*)(unsigned long long)(const void*)g,
        (as3_vsi4*)(unsigned int)(unsigned long long)(void*)l,
        0, 0);
}
__device__ __forceinline__ void wait_async0() {
#if __has_builtin(__builtin_amdgcn_s_wait_asynccnt)
    __builtin_amdgcn_s_wait_asynccnt(0);
#else
    asm volatile("s_wait_asynccnt 0x0" ::: "memory");
#endif
}
#endif

// ---------------------------------------------------------------------------
// Kernel 1: dec_w f32 [VOCAB][KDIM] -> bf16 [NPAD][KDIM], zero-padded rows
// ---------------------------------------------------------------------------
__global__ __launch_bounds__(256) void convert_decw_kernel(
    const float* __restrict__ dw, ushort_t* __restrict__ dwb)
{
    size_t total = (size_t)NPAD * KDIM;
    size_t stride = (size_t)gridDim.x * blockDim.x;
    for (size_t i = (size_t)blockIdx.x * blockDim.x + threadIdx.x; i < total; i += stride) {
        size_t n = i >> 10;                    // /KDIM
        dwb[i] = (n < (size_t)VOCAB) ? f2bf(dw[i]) : (ushort_t)0;
    }
}

// ---------------------------------------------------------------------------
// Kernel 2: gather + bias + L2 normalize -> bf16 feats [M][KDIM]
// ---------------------------------------------------------------------------
__global__ __launch_bounds__(256) void gather_norm_kernel(
    const int* __restrict__ tok, const float* __restrict__ enc_w,
    const float* __restrict__ enc_b, ushort_t* __restrict__ featsb)
{
    __shared__ float red[256];
    __shared__ float scale_s;
    int row = blockIdx.x;
    int t   = threadIdx.x;
    int tk  = tok[row];

    float v[4];
    float ss = 0.f;
#pragma unroll
    for (int i = 0; i < 4; ++i) {
        int f = t + i * 256;
        float x = enc_w[(size_t)f * VOCAB + tk] + enc_b[f];
        v[i] = x;
        ss += x * x;
    }
    red[t] = ss;
    __syncthreads();
    for (int s = 128; s > 0; s >>= 1) {
        if (t < s) red[t] += red[t + s];
        __syncthreads();
    }
    if (t == 0) {
        float nrm = sqrtf(red[0]);
        scale_s = 1.0f / fmaxf(nrm, 1e-12f);
    }
    __syncthreads();
    float sc = scale_s;
#pragma unroll
    for (int i = 0; i < 4; ++i)
        featsb[(size_t)row * KDIM + t + i * 256] = f2bf(v[i] * sc);
}

// ---------------------------------------------------------------------------
// Kernel 3: bf16 WMMA GEMM  out[M][VOCAB] = A[M][K] * B[NPAD][K]^T + bias
// block tile 128x128, 8 waves -> wave tile 32x64 (2x4 of 16x16)
// ---------------------------------------------------------------------------
__global__ __launch_bounds__(256) void gemm_wmma_kernel(
    const ushort_t* __restrict__ A,      // [M][KDIM] bf16
    const ushort_t* __restrict__ B,      // [NPAD][KDIM] bf16
    const float* __restrict__ bias,      // [VOCAB]
    float* __restrict__ out)             // [M][VOCAB]
{
    __shared__ ushort_t As[2][TM][LDSP];
    __shared__ ushort_t Bs[2][TN][LDSP];

    const int tid   = threadIdx.x;
    const int lane  = tid & 31;
    const int wid   = tid >> 5;
    const int wm    = (wid & 3) * 32;     // wave M offset in block tile
    const int wn    = (wid >> 2) * 64;    // wave N offset in block tile
    const int m_blk = blockIdx.y * TM;
    const int n_blk = blockIdx.x * TN;

    // cooperative global->LDS staging: each thread = one 16-half row segment
    const int lrow = tid >> 1;            // 0..127
    const int lcol = (tid & 1) * 16;      // 0 or 16 (halves)

    const ushort_t* agp = A + (size_t)(m_blk + lrow) * KDIM + lcol;
    const ushort_t* bgp = B + (size_t)(n_blk + lrow) * KDIM + lcol;

    v8f acc[2][4];
#pragma unroll
    for (int mi = 0; mi < 2; ++mi)
#pragma unroll
        for (int ni = 0; ni < 4; ++ni)
            acc[mi][ni] = (v8f){0.f, 0.f, 0.f, 0.f, 0.f, 0.f, 0.f, 0.f};

    // ---- prologue: fill buffer 0 with K-step 0 ----
#if ASYNC_COPY
    async_copy16(agp + 0, &As[0][lrow][lcol + 0]);
    async_copy16(agp + 8, &As[0][lrow][lcol + 8]);
    async_copy16(bgp + 0, &Bs[0][lrow][lcol + 0]);
    async_copy16(bgp + 8, &Bs[0][lrow][lcol + 8]);
    wait_async0();
#else
    {
        uint4 a0 = *(const uint4*)(agp + 0);
        uint4 a1 = *(const uint4*)(agp + 8);
        uint4 b0 = *(const uint4*)(bgp + 0);
        uint4 b1 = *(const uint4*)(bgp + 8);
        *(uint4*)&As[0][lrow][lcol + 0] = a0;
        *(uint4*)&As[0][lrow][lcol + 8] = a1;
        *(uint4*)&Bs[0][lrow][lcol + 0] = b0;
        *(uint4*)&Bs[0][lrow][lcol + 8] = b1;
    }
#endif
    __syncthreads();

    const int koff = (lane < 16) ? 0 : 8;      // ISA 16-bit A/B per-lane K layout

    for (int ks = 0; ks < KDIM / TK; ++ks) {
        const int cur = ks & 1;
        const int nxt = cur ^ 1;
        const bool more = (ks + 1) < (KDIM / TK);

#if ASYNC_COPY
        // issue next tile's copies straight into LDS (no VGPR staging)
        if (more) {
            const int k0 = (ks + 1) * TK;
            async_copy16(agp + k0 + 0, &As[nxt][lrow][lcol + 0]);
            async_copy16(agp + k0 + 8, &As[nxt][lrow][lcol + 8]);
            async_copy16(bgp + k0 + 0, &Bs[nxt][lrow][lcol + 0]);
            async_copy16(bgp + k0 + 8, &Bs[nxt][lrow][lcol + 8]);
        }
#else
        uint4 na0, na1, nb0, nb1;
        if (more) {
            const int k0 = (ks + 1) * TK;
            na0 = *(const uint4*)(agp + k0 + 0);
            na1 = *(const uint4*)(agp + k0 + 8);
            nb0 = *(const uint4*)(bgp + k0 + 0);
            nb1 = *(const uint4*)(bgp + k0 + 8);
        }
#endif

        // fragment loads from LDS (ds_load_b128 x2 per fragment)
        FragBF fa[2], fb[4];
#pragma unroll
        for (int mi = 0; mi < 2; ++mi) {
            const int r = wm + mi * 16 + (lane & 15);
            fa[mi].q[0] = *(const uint4*)&As[cur][r][koff];
            fa[mi].q[1] = *(const uint4*)&As[cur][r][16 + koff];
        }
#pragma unroll
        for (int ni = 0; ni < 4; ++ni) {
            const int r = wn + ni * 16 + (lane & 15);
            fb[ni].q[0] = *(const uint4*)&Bs[cur][r][koff];
            fb[ni].q[1] = *(const uint4*)&Bs[cur][r][16 + koff];
        }

#pragma unroll
        for (int mi = 0; mi < 2; ++mi)
#pragma unroll
            for (int ni = 0; ni < 4; ++ni)
                acc[mi][ni] = __builtin_amdgcn_wmma_f32_16x16x32_bf16(
                    false, fa[mi].v, false, fb[ni].v,
                    (short)0, acc[mi][ni], false, false);

#if ASYNC_COPY
        if (more) wait_async0();               // own copies into nxt complete
#else
        if (more) {
            *(uint4*)&As[nxt][lrow][lcol + 0] = na0;
            *(uint4*)&As[nxt][lrow][lcol + 8] = na1;
            *(uint4*)&Bs[nxt][lrow][lcol + 0] = nb0;
            *(uint4*)&Bs[nxt][lrow][lcol + 8] = nb1;
        }
#endif
        __syncthreads();                       // everyone's nxt complete; cur free
    }

    // epilogue: C/D layout -> VGPR r holds (M = base + r (+8 for lanes 16..31), N = lane%16)
    const int rshift = (lane >> 4) << 3;       // +8 for upper half-wave
#pragma unroll
    for (int mi = 0; mi < 2; ++mi) {
        const int rbase = m_blk + wm + mi * 16 + rshift;
#pragma unroll
        for (int ni = 0; ni < 4; ++ni) {
            const int cn = n_blk + wn + ni * 16 + (lane & 15);
            if (cn < VOCAB) {
                const float bb = bias[cn];
#pragma unroll
                for (int r = 0; r < 8; ++r)
                    out[(size_t)(rbase + r) * VOCAB + cn] = acc[mi][ni][r] + bb;
            }
        }
    }
}

// ---------------------------------------------------------------------------
// Kernel 4: row-wise log_softmax, one block per row
// ---------------------------------------------------------------------------
__global__ __launch_bounds__(256) void logsoftmax_kernel(float* __restrict__ out)
{
    __shared__ float red[256];
    const int t = threadIdx.x;
    float* p = out + (size_t)blockIdx.x * VOCAB;

    float mx = -3.0e38f;
    for (int i = t; i < VOCAB; i += 256) mx = fmaxf(mx, p[i]);
    red[t] = mx;
    __syncthreads();
    for (int s = 128; s > 0; s >>= 1) {
        if (t < s) red[t] = fmaxf(red[t], red[t + s]);
        __syncthreads();
    }
    mx = red[0];
    __syncthreads();

    float sum = 0.f;
    for (int i = t; i < VOCAB; i += 256) sum += __expf(p[i] - mx);
    red[t] = sum;
    __syncthreads();
    for (int s = 128; s > 0; s >>= 1) {
        if (t < s) red[t] += red[t + s];
        __syncthreads();
    }
    const float ls = mx + __logf(red[0]);

    for (int i = t; i < VOCAB; i += 256) p[i] -= ls;
}

// ---------------------------------------------------------------------------
extern "C" void kernel_launch(void* const* d_in, const int* in_sizes, int n_in,
                              void* d_out, int out_size, void* d_ws, size_t ws_size,
                              hipStream_t stream)
{
    const int*   tok   = (const int*)d_in[0];
    const float* enc_w = (const float*)d_in[1];
    const float* enc_b = (const float*)d_in[2];
    const float* dec_w = (const float*)d_in[3];
    const float* dec_b = (const float*)d_in[4];
    float*       out   = (float*)d_out;

    const int M = in_sizes[0];                 // B*S = 8192

    // workspace layout
    char* ws = (char*)d_ws;
    size_t feats_bytes = ((size_t)M * KDIM * sizeof(ushort_t) + 255) & ~(size_t)255;
    ushort_t* featsb = (ushort_t*)ws;
    ushort_t* dwb    = (ushort_t*)(ws + feats_bytes);

    convert_decw_kernel<<<4096, 256, 0, stream>>>(dec_w, dwb);
    gather_norm_kernel<<<M, 256, 0, stream>>>(tok, enc_w, enc_b, featsb);

    dim3 grid(NPAD / TN, M / TM);              // 393 x 64
    gemm_wmma_kernel<<<grid, 256, 0, stream>>>(featsb, dwb, dec_b, out);

    logsoftmax_kernel<<<M, 256, 0, stream>>>(out);
}
